// SensorWiseMaskEncoder_15539191677824
// MI455X (gfx1250) — matrix operand order repaired and checked by reading, not compiled, not run on hardware
//
#include <hip/hip_runtime.h>
#include <hip/hip_bf16.h>
#include <stdint.h>

// ---------------------------------------------------------------------------
// Problem constants (from reference): B=256, F=512, D=768, E=512, NM=256
// ---------------------------------------------------------------------------
#define B_   256
#define F_   512
#define D_   768
#define E_   512
#define H_   384    // D/2 : per-sensor input width (K of each GEMM)
#define EH_  256    // E/2 : per-sensor output width
#define NM_  256
#define UM_  256    // F - NM
#define M_TOTAL (B_ * F_)   // 131072 rows

// GEMM tiling
#define MT  64      // rows per block
#define NT  128     // cols per block (always within one sensor: E/NT = 4 tiles)
#define KT  32      // K per WMMA step (bf16 shape 16x16x32)
#define LDA 40      // LDS halfword stride (80B rows -> conflict-free, 16B aligned)

#define WSPL_HALF_BYTES (E_ * H_ * 2)   // bytes of one split-weight plane (hi or lo)

typedef __attribute__((ext_vector_type(16))) __bf16 v16bf;
typedef __attribute__((ext_vector_type(8)))  float  v8f;

struct Frag32 { uint4 lo, hi; };   // 32 bytes == one v16bf fragment

__device__ __forceinline__ unsigned short f32_to_bf16_rne(float x) {
    unsigned int u = __float_as_uint(x);
    unsigned int r = u + 0x7FFFu + ((u >> 16) & 1u);   // round-to-nearest-even
    return (unsigned short)(r >> 16);
}
__device__ __forceinline__ float bf16_to_f32(unsigned short h) {
    return __uint_as_float(((unsigned int)h) << 16);
}
__device__ __forceinline__ v16bf load_frag(const unsigned short* p0, const unsigned short* p1) {
    Frag32 t;
    t.lo = *(const uint4*)p0;
    t.hi = *(const uint4*)p1;
    return __builtin_bit_cast(v16bf, t);
}
__device__ __forceinline__ unsigned lds_off32(const void* p) {
    // generic pointer to LDS: low 32 bits are the LDS byte address
    return (unsigned)(uintptr_t)p;
}

// ---------------------------------------------------------------------------
// Kernel 0: one-time weight split: acc_w|gyro_w -> transposed bf16 hi/lo
//           planes  whiT[n][k], wloT[n][k]  (n in [0,512), k in [0,384)).
// ---------------------------------------------------------------------------
__global__ __launch_bounds__(384) void wsplit_kernel(
    const float* __restrict__ acc_w, const float* __restrict__ gyro_w,
    unsigned short* __restrict__ whiT, unsigned short* __restrict__ wloT)
{
    const int n = blockIdx.x;      // 0..511 (output column, both sensors)
    const int k = threadIdx.x;     // 0..383
    const float w = (n < EH_) ? acc_w[(size_t)k * EH_ + n]
                              : gyro_w[(size_t)k * EH_ + (n - EH_)];
    const unsigned short h = f32_to_bf16_rne(w);
    whiT[(size_t)n * H_ + k] = h;
    wloT[(size_t)n * H_ + k] = f32_to_bf16_rne(w - bf16_to_f32(h));
}

// ---------------------------------------------------------------------------
// Kernel 1: split-bf16 (bf16x3) GEMM  patch = framed @ blockdiag(acc_w,gyro_w)
//           + bias + pos_emb fused epilogue.
// Double-buffered pipeline: W tiles arrive via async global->LDS DMA
// (ASYNCcnt), A tiles are prefetched to VGPRs, split, and stored to LDS.
// ---------------------------------------------------------------------------
__device__ __forceinline__ void load_a_regs(
    const float* __restrict__ framed, int tid, int mBase, int sOff, int k0,
    float4& a0, float4& a1)
{
    const int i0 = tid, i1 = 256 + tid;
    a0 = *(const float4*)&framed[(size_t)(mBase + (i0 >> 3)) * D_ + sOff + k0 + ((i0 & 7) << 2)];
    a1 = *(const float4*)&framed[(size_t)(mBase + (i1 >> 3)) * D_ + sOff + k0 + ((i1 & 7) << 2)];
}

__device__ __forceinline__ void store_a_lds(
    unsigned short* __restrict__ ahi, unsigned short* __restrict__ alo,
    int tid, const float4& a0, const float4& a1)
{
#pragma unroll
    for (int j = 0; j < 2; ++j) {
        const int idx = j * 256 + tid;
        const int rr  = idx >> 3;
        const int kk  = (idx & 7) << 2;
        const float4 v = j ? a1 : a0;
        const float f[4] = {v.x, v.y, v.z, v.w};
        unsigned short h[4], l[4];
#pragma unroll
        for (int c = 0; c < 4; ++c) {
            h[c] = f32_to_bf16_rne(f[c]);
            l[c] = f32_to_bf16_rne(f[c] - bf16_to_f32(h[c]));
        }
        uint2 ph, pl;
        ph.x = (unsigned)h[0] | ((unsigned)h[1] << 16);
        ph.y = (unsigned)h[2] | ((unsigned)h[3] << 16);
        pl.x = (unsigned)l[0] | ((unsigned)l[1] << 16);
        pl.y = (unsigned)l[2] | ((unsigned)l[3] << 16);
        *(uint2*)&ahi[rr * LDA + kk] = ph;
        *(uint2*)&alo[rr * LDA + kk] = pl;
    }
}

// Async DMA of one [NT x KT] bf16 W tile pair (hi+lo) into padded LDS rows.
// 1024 16B chunks / 256 threads = 4 per thread.  Per-lane LDS address keeps
// the 80B padded row stride so B-fragment ds_load_b128 stay conflict-free.
__device__ __forceinline__ void issue_w_async(
    const unsigned short* __restrict__ wsplit, int tid, int cBase, int k0,
    unsigned whB, unsigned wlB)
{
#pragma unroll
    for (int j = 0; j < 4; ++j) {
        const int idx = j * 256 + tid;
        const int sel = idx >> 9;        // 0: hi plane, 1: lo plane
        const int rem = idx & 511;
        const int n   = rem >> 2;        // 0..127 local column
        const int c   = rem & 3;         // 16B chunk within 64B row slice
        const unsigned lds  = (sel ? wlB : whB) + (unsigned)(n * (LDA * 2) + c * 16);
        const unsigned voff = (unsigned)((cBase + n) * (H_ * 2) + k0 * 2 + c * 16
                                         + sel * WSPL_HALF_BYTES);
        asm volatile("global_load_async_to_lds_b128 %0, %1, %2 offset:0"
                     :: "v"(lds), "v"(voff), "s"(wsplit) : "memory");
    }
}

__global__ __launch_bounds__(256) void gemm_patch_kernel(
    const float* __restrict__ framed,
    const unsigned short* __restrict__ wsplit,   // whiT plane then wloT plane
    const float* __restrict__ acc_b, const float* __restrict__ gyro_b,
    const float* __restrict__ pos_emb,
    float* __restrict__ patch)
{
    __shared__ alignas(16) unsigned short Ahi[2][MT * LDA];
    __shared__ alignas(16) unsigned short Alo[2][MT * LDA];
    __shared__ alignas(16) unsigned short Wh [2][NT * LDA];
    __shared__ alignas(16) unsigned short Wl [2][NT * LDA];

    const int tid   = threadIdx.x;
    const int bid   = blockIdx.x;
    const int nTile = bid & 3;            // E/NT = 4 col tiles
    const int mTile = bid >> 2;
    const int sensor = nTile >> 1;        // 0: acc, 1: gyro
    const int cBase  = nTile * NT;        // global col base [0,512)
    const int mBase  = mTile * MT;
    const int sOff   = sensor * H_;

    const float* bPtr = sensor ? gyro_b : acc_b;

    const int lane  = tid & 31;           // wave32
    const int wave  = tid >> 5;           // 8 waves
    const int r     = lane & 15;
    const int half  = lane >> 4;
    const int mSub  = wave & 3;           // wave's 16-row subtile (of 4)
    const int nGrp  = wave >> 2;          // wave's 64-col group (of 2)

    const unsigned whB[2] = {lds_off32(&Wh[0][0]), lds_off32(&Wh[1][0])};
    const unsigned wlB[2] = {lds_off32(&Wl[0][0]), lds_off32(&Wl[1][0])};

    v8f acc[4] = {};                      // 4 x (16x16 f32) accumulators
    float4 aReg0, aReg1;

    // ---- prologue: stage 0 ----
    load_a_regs(framed, tid, mBase, sOff, 0, aReg0, aReg1);
    issue_w_async(wsplit, tid, cBase, 0, whB[0], wlB[0]);
    store_a_lds(&Ahi[0][0], &Alo[0][0], tid, aReg0, aReg1);
    asm volatile("s_wait_asynccnt 0x0" ::: "memory");
    __syncthreads();

    for (int kt = 0; kt < H_ / KT; ++kt) {
        const int buf = kt & 1;
        const int nxt = buf ^ 1;
        const bool more = (kt + 1) < H_ / KT;

        // ---- prefetch stage kt+1 (A to regs, W via async DMA) ----
        if (more) {
            load_a_regs(framed, tid, mBase, sOff, (kt + 1) * KT, aReg0, aReg1);
            issue_w_async(wsplit, tid, cBase, (kt + 1) * KT, whB[nxt], wlB[nxt]);
        }

        // ---- compute stage kt: 12 WMMAs per wave ----
        // A layout (ISA 7.12.2, 16-bit A 16x32): lane(r,half) holds row r,
        // K = half*8+[0..8) in v[0..3], K = 16+half*8+[0..8) in v[4..7].
        const int aOff = (mSub * 16 + r) * LDA + half * 8;
        const v16bf aH = load_frag(&Ahi[buf][aOff], &Ahi[buf][aOff + 16]);
        const v16bf aL = load_frag(&Alo[buf][aOff], &Alo[buf][aOff + 16]);

#pragma unroll
        for (int ns = 0; ns < 4; ++ns) {
            // B layout: lane(r,half) holds column r, K = half*8+[0..8), 16+half*8+[0..8).
            const int bOff = ((nGrp * 4 + ns) * 16 + r) * LDA + half * 8;
            const v16bf bH = load_frag(&Wh[buf][bOff], &Wh[buf][bOff + 16]);
            const v16bf bL = load_frag(&Wl[buf][bOff], &Wl[buf][bOff + 16]);
            // bf16x3: a*b ~= aH*bH + aH*bL + aL*bH  (fp32 accumulate)
            acc[ns] = __builtin_amdgcn_wmma_f32_16x16x32_bf16(
                false, aH, false, bH, (short)0, acc[ns], false, false);
            acc[ns] = __builtin_amdgcn_wmma_f32_16x16x32_bf16(
                false, aH, false, bL, (short)0, acc[ns], false, false);
            acc[ns] = __builtin_amdgcn_wmma_f32_16x16x32_bf16(
                false, aL, false, bH, (short)0, acc[ns], false, false);
        }

        // ---- publish stage kt+1 ----
        if (more) {
            store_a_lds(&Ahi[nxt][0], &Alo[nxt][0], tid, aReg0, aReg1);
            asm volatile("s_wait_asynccnt 0x0" ::: "memory");
        }
        __syncthreads();
    }

    // ---- epilogue: + bias + pos_emb, store patch ----
    // C/D layout: VGPR v -> M = v + 8*half, N = lane&15.
#pragma unroll
    for (int ns = 0; ns < 4; ++ns) {
        const int col  = cBase + (nGrp * 4 + ns) * 16 + r;
        const float bias = bPtr[col & (EH_ - 1)];
#pragma unroll
        for (int v = 0; v < 8; ++v) {
            const int row = mBase + mSub * 16 + half * 8 + v;
            const float val = acc[ns][v] + bias
                            + pos_emb[(size_t)(row & (F_ - 1)) * E_ + col];
            patch[(size_t)row * E_ + col] = val;
        }
    }
}

// ---------------------------------------------------------------------------
// Kernel 2: mask-token projection (one [1,512] vector) into workspace.
// ---------------------------------------------------------------------------
__global__ __launch_bounds__(512) void mtproj_kernel(
    const float* __restrict__ mask_token,
    const float* __restrict__ acc_w,  const float* __restrict__ acc_b,
    const float* __restrict__ gyro_w, const float* __restrict__ gyro_b,
    float* __restrict__ mt_proj)
{
    const int e  = threadIdx.x;          // 0..511
    const int s  = e >> 8;
    const int eo = e & 255;
    const float* w   = s ? gyro_w : acc_w;
    const float* tok = mask_token + s * H_;
    float sum = (s ? gyro_b : acc_b)[eo];
    for (int k = 0; k < H_; ++k)
        sum = fmaf(tok[k], w[(size_t)k * EH_ + eo], sum);
    mt_proj[e] = sum;
}

// ---------------------------------------------------------------------------
// Kernel 3: assemble encoder_input = concat(gather(patch), mt_proj + pos_emb).
// ---------------------------------------------------------------------------
__global__ __launch_bounds__(256) void assemble_kernel(
    const float* __restrict__ patch,
    const int* __restrict__ mask_indices,
    const int* __restrict__ unmask_indices,
    const float* __restrict__ pos_emb,
    const float* __restrict__ mt_proj,
    float* __restrict__ enc)
{
    const int tid = threadIdx.x;
    const int rowLocal = tid >> 7;                 // 2 rows per block
    const int q  = (tid & 127) << 2;               // float index within row
    const int row = blockIdx.x * 2 + rowLocal;     // [0, B*F)
    const int b = row >> 9;                        // row / F
    const int j = row & (F_ - 1);                  // row % F
    float4 out;
    if (j < UM_) {
        const int src = unmask_indices[b * UM_ + j];
        out = *(const float4*)&patch[(size_t)(b * F_ + src) * E_ + q];
    } else {
        const int mi = mask_indices[b * NM_ + (j - UM_)];
        const float4 p = *(const float4*)&pos_emb[(size_t)mi * E_ + q];
        const float4 t = *(const float4*)&mt_proj[q];
        out = make_float4(p.x + t.x, p.y + t.y, p.z + t.z, p.w + t.w);
    }
    *(float4*)&enc[(size_t)row * E_ + q] = out;
}

// ---------------------------------------------------------------------------
// Kernel 4: emit mask_indices (third tuple output) in the output dtype.
// ---------------------------------------------------------------------------
__global__ __launch_bounds__(256) void maskcast_kernel(
    const int* __restrict__ mi, float* __restrict__ out)
{
    const int i = blockIdx.x * blockDim.x + threadIdx.x;
    out[i] = (float)mi[i];
}

// ---------------------------------------------------------------------------
extern "C" void kernel_launch(void* const* d_in, const int* in_sizes, int n_in,
                              void* d_out, int out_size, void* d_ws, size_t ws_size,
                              hipStream_t stream)
{
    (void)in_sizes; (void)n_in; (void)out_size; (void)ws_size;

    const float* framed     = (const float*)d_in[0];
    const int*   mask_idx   = (const int*)  d_in[1];
    const int*   unmask_idx = (const int*)  d_in[2];
    const float* acc_w      = (const float*)d_in[3];
    const float* acc_b      = (const float*)d_in[4];
    const float* gyro_w     = (const float*)d_in[5];
    const float* gyro_b     = (const float*)d_in[6];
    const float* pos_emb    = (const float*)d_in[7];
    const float* mask_token = (const float*)d_in[8];

    float* enc   = (float*)d_out;                        // [B,F,E]
    float* patch = enc   + (size_t)B_ * F_ * E_;         // [B,F,E]
    float* maskf = patch + (size_t)B_ * F_ * E_;         // [B,NM] as float
    float* mtp   = (float*)d_ws;                         // [E] scratch

    // Split-weight planes parked at the start of the enc region: the GEMM
    // (its only reader) completes before assemble_kernel overwrites enc.
    unsigned short* whiT = (unsigned short*)enc;
    unsigned short* wloT = whiT + (size_t)E_ * H_;

    wsplit_kernel<<<E_, H_, 0, stream>>>(acc_w, gyro_w, whiT, wloT);
    mtproj_kernel<<<1, 512, 0, stream>>>(mask_token, acc_w, acc_b,
                                         gyro_w, gyro_b, mtp);
    gemm_patch_kernel<<<(M_TOTAL / MT) * (E_ / NT), 256, 0, stream>>>(
        framed, whiT, acc_b, gyro_b, pos_emb, patch);
    assemble_kernel<<<M_TOTAL / 2, 256, 0, stream>>>(
        patch, mask_idx, unmask_idx, pos_emb, mtp, enc);
    maskcast_kernel<<<(B_ * NM_) / 256, 256, 0, stream>>>(mask_idx, maskf);
}
